// QuickReadoutNet_45037027066073
// MI455X (gfx1250) — matrix operand order repaired and checked by reading, not compiled, not run on hardware
//
#include <hip/hip_runtime.h>
#include <hip/hip_bf16.h>

// ---------------------------------------------------------------------------
// QuickReadoutNet on MI455X (gfx1250): bf16 WMMA path.
//  - 64-edge M-tiling (MT=4) per workgroup
//  - kt-outer register-blocked layers: 16 C-frags live, 16 WMMAs per
//    (4 A-load + 4 B-load) step
//  - async global->LDS gather (ASYNCcnt) for the edge feature scatter
// ---------------------------------------------------------------------------

typedef __attribute__((ext_vector_type(16))) __bf16 v16bf;
typedef __attribute__((ext_vector_type(8)))  __bf16 v8bf;
typedef __attribute__((ext_vector_type(8)))  float  v8f;

#define N_NODES 50000
#define W_FEAT  256
#define DHID    512
#define NU      16
#define NB      32
#define E_EDGES 262144
#define MT      4   // 16-edge M-tiles per workgroup in the edge kernel

__device__ __forceinline__ unsigned short f2bf(float f) {
  unsigned u = __float_as_uint(f);
  u += 0x7FFFu + ((u >> 16) & 1u);        // round-to-nearest-even
  return (unsigned short)(u >> 16);
}

__device__ __forceinline__ v8f wmma_bf16(v16bf a, v16bf b, v8f c) {
  // (neg_a, A, neg_b, B, c_mod, C, reuse_a, reuse_b)
  return __builtin_amdgcn_wmma_f32_16x16x32_bf16(false, a, false, b, (short)0, c,
                                                 false, false);
}

// f32 -> bf16 conversion of objs (12.8M elems, 4/thread)
__global__ void k_cvt_objs(const float* __restrict__ src,
                           unsigned short* __restrict__ dst, int n) {
  int i = (blockIdx.x * blockDim.x + threadIdx.x) * 4;
  if (i + 3 < n) {
    float4 v = *(const float4*)(src + i);
    ushort4 o;
    o.x = f2bf(v.x); o.y = f2bf(v.y); o.z = f2bf(v.z); o.w = f2bf(v.w);
    *(ushort4*)(dst + i) = o;
  }
}

// Pack a (K x Nc) row-major f32 weight matrix into per-(32x16)-block bf16
// B-fragment layout: block = (k/32)*(Nc/16) + n/16 ; within a block, lane
// (n&15)+16*(((k&31)>>3)&1) holds element i=(k&7)+8*((k&31)>>4)  -> each lane reads 32B.
__global__ void k_pack_w(const float* __restrict__ src,
                         unsigned short* __restrict__ dst, int K, int Nc) {
  int id = blockIdx.x * blockDim.x + threadIdx.x;
  if (id >= K * Nc) return;
  int k = id / Nc, n = id % Nc;
  int blk  = (k >> 5) * (Nc >> 4) + (n >> 4);
  int kk   = k & 31;
  int lane = (n & 15) + (((kk >> 3) & 1) << 4);
  int i    = (kk & 7) + ((kk >> 4) << 3);
  dst[blk * 512 + lane * 16 + i] = f2bf(src[id]);
}

// Pack relu(C-fragment) value at (m, n) into the 16-bit A-fragment LDS layout
// for the next layer (n becomes K). A layout: lanes 0-15 hold K {0..7,16..23}
// of a 32-chunk, lanes 16-31 hold K {8..15,24..31}.
__device__ __forceinline__ void pack_a_elem(unsigned short* ldsDst, int m,
                                            int n, float v) {
  int kt = n >> 5, kk = n & 31;
  int lh = ((kk & 15) >= 8) ? 1 : 0;
  int wl = m + 16 * lh;
  int ii = (kk < 16) ? (kk - 8 * lh) : (kk - 8 - 8 * lh);
  ldsDst[kt * 512 + wl * 16 + ii] = f2bf(v);
}

// ---------------------------------------------------------------------------
// Node MLP: one 16-node tile per 128-thread (4-wave) block.
// ---------------------------------------------------------------------------
__global__ void __launch_bounds__(128)
k_node_mlp(const unsigned short* __restrict__ objsB,
           const unsigned short* __restrict__ pw1,  // fu_w1 packed: 8kT x 32nT
           const float* __restrict__ b1,
           const unsigned short* __restrict__ pw2,  // fu_w2 packed: 16kT x 1nT
           const float* __restrict__ b2,
           float* __restrict__ z) {
  __shared__ __attribute__((aligned(32))) unsigned short ldsH[16 * 512];
  const int tile  = blockIdx.x;
  const int lane  = threadIdx.x & 31;
  const int wid   = threadIdx.x >> 5;
  const int nLoc  = lane & 15;
  const int hiH   = lane >> 4;
  const int mBase = hiH * 8;   // C-fragment row base
  const int kBase = hiH * 8;   // A-fragment K base
  const unsigned short* aRow = objsB + (size_t)(tile * 16 + nLoc) * W_FEAT;

  // layer 1: each wave handles 8 of the 32 N-tiles
  for (int q = 0; q < 8; ++q) {
    int nt = wid * 8 + q;
    float bias = b1[nt * 16 + nLoc];
    v8f c = {bias, bias, bias, bias, bias, bias, bias, bias};
#pragma unroll
    for (int kt = 0; kt < 8; ++kt) {
      v8bf lo = *(const v8bf*)(aRow + kt * 32 + kBase);
      v8bf hi = *(const v8bf*)(aRow + kt * 32 + 16 + kBase);
      v16bf a;
#pragma unroll
      for (int j = 0; j < 8; ++j) { a[j] = lo[j]; a[8 + j] = hi[j]; }
      v16bf b = *(const v16bf*)(pw1 + (kt * 32 + nt) * 512 + lane * 16);
      c = wmma_bf16(a, b, c);
    }
#pragma unroll
    for (int r = 0; r < 8; ++r)
      pack_a_elem(ldsH, r + mBase, nt * 16 + nLoc, fmaxf(c[r], 0.0f));
  }
  __syncthreads();

  // layer 2 (16x512 @ 512x16) by wave 0
  if (wid == 0) {
    float bias = b2[nLoc];
    v8f c = {bias, bias, bias, bias, bias, bias, bias, bias};
#pragma unroll
    for (int kt = 0; kt < 16; ++kt) {
      v16bf a = *(const v16bf*)(ldsH + kt * 512 + lane * 16);
      v16bf b = *(const v16bf*)(pw2 + kt * 512 + lane * 16);
      c = wmma_bf16(a, b, c);
    }
#pragma unroll
    for (int r = 0; r < 8; ++r)
      z[(size_t)(tile * 16 + r + mBase) * NU + nLoc] = c[r];
  }
}

// ---------------------------------------------------------------------------
// Edge MLP: 64 edges (4 x 16-edge M-tiles) per 256-thread (8-wave) block.
// Each wave owns 4 N-tiles x 4 M-tiles (16 C-frags live). Per kt step:
// 4 A LDS loads + 4 B global loads feed 16 WMMAs.
// ---------------------------------------------------------------------------
__global__ void __launch_bounds__(256)
k_edge_mlp(const unsigned short* __restrict__ objsB,
           const long long* __restrict__ ei,        // int64 (2, E)
           const unsigned short* __restrict__ pw1,  // fb_w1 packed: 16kT x 32nT
           const float* __restrict__ b1,
           const unsigned short* __restrict__ pw2,  // fb_w2 packed: 16kT x 32nT
           const float* __restrict__ b2,
           const unsigned short* __restrict__ pw3,  // fb_w3 packed: 16kT x 2nT
           const float* __restrict__ b3,
           float* __restrict__ y) {
  __shared__ __attribute__((aligned(32))) unsigned short ldsA[MT][16 * 512];
  __shared__ __attribute__((aligned(32))) unsigned short ldsB[MT][16 * 512];
  const int blk   = blockIdx.x;            // group of 64 edges
  const int t     = threadIdx.x;
  const int lane  = t & 31;
  const int wid   = t >> 5;
  const int nLoc  = lane & 15;
  const int mBase = (lane >> 4) * 8;

  // S0: async gather of x1|x2 rows (bf16) straight into packed-A LDS layout.
  // thread = (edge m = t>>4, 32-feature chunk = t&15); chunk == kTile.
  // 64B of a node row scatter as 4x16B: k0..7 ->(m,i0..7), k8..15 ->(m+16,
  // i0..7), k16..23 ->(m,i8..15), k24..31 ->(m+16,i8..15).
  {
    const int m = t >> 4;
    const int chunk = t & 15;
#pragma unroll
    for (int mt = 0; mt < MT; ++mt) {
      const int e = blk * (16 * MT) + mt * 16 + m;
      const int s = (int)ei[e];
      const int d = (int)ei[E_EDGES + e];
      const int row = (chunk < 8) ? s : d;
      unsigned long long g = (unsigned long long)(size_t)(
          objsB + (size_t)row * W_FEAT + (chunk & 7) * 32);
      unsigned short* dstBase = ldsA[mt] + chunk * 512;
      unsigned l0 = (unsigned)(size_t)(dstBase + m * 16);
      unsigned l1 = (unsigned)(size_t)(dstBase + (m + 16) * 16);
      unsigned l2 = (unsigned)(size_t)(dstBase + m * 16 + 8);
      unsigned l3 = (unsigned)(size_t)(dstBase + (m + 16) * 16 + 8);
      asm volatile(
          "global_load_async_to_lds_b128 %0, %4, off\n\t"
          "global_load_async_to_lds_b128 %1, %4, off offset:16\n\t"
          "global_load_async_to_lds_b128 %2, %4, off offset:32\n\t"
          "global_load_async_to_lds_b128 %3, %4, off offset:48"
          :
          : "v"(l0), "v"(l1), "v"(l2), "v"(l3), "v"(g)
          : "memory");
    }
    asm volatile("s_wait_asynccnt 0x0" ::: "memory");
  }
  __syncthreads();

  // layer 1: kt-outer register blocking, ldsA -> relu -> ldsB
  {
    v8f c[4][MT];
#pragma unroll
    for (int qn = 0; qn < 4; ++qn) {
      float bias = b1[(wid * 4 + qn) * 16 + nLoc];
#pragma unroll
      for (int mt = 0; mt < MT; ++mt)
        c[qn][mt] = (v8f){bias, bias, bias, bias, bias, bias, bias, bias};
    }
#pragma unroll 4
    for (int kt = 0; kt < 16; ++kt) {
      v16bf a[MT];
#pragma unroll
      for (int mt = 0; mt < MT; ++mt)
        a[mt] = *(const v16bf*)(ldsA[mt] + kt * 512 + lane * 16);
#pragma unroll
      for (int qn = 0; qn < 4; ++qn) {
        const unsigned short* bp =
            pw1 + (kt * 32 + wid * 4 + qn) * 512 + lane * 16;
        v16bf b = *(const v16bf*)bp;
        if (qn == 0 && kt < 15) __builtin_prefetch(bp + 32 * 512, 0, 0);
#pragma unroll
        for (int mt = 0; mt < MT; ++mt)
          c[qn][mt] = wmma_bf16(a[mt], b, c[qn][mt]);
      }
    }
#pragma unroll
    for (int qn = 0; qn < 4; ++qn)
#pragma unroll
      for (int mt = 0; mt < MT; ++mt)
#pragma unroll
        for (int r = 0; r < 8; ++r)
          pack_a_elem(ldsB[mt], r + mBase, (wid * 4 + qn) * 16 + nLoc,
                      fmaxf(c[qn][mt][r], 0.0f));
  }
  __syncthreads();

  // layer 2: same shape, ldsB -> relu -> ldsA
  {
    v8f c[4][MT];
#pragma unroll
    for (int qn = 0; qn < 4; ++qn) {
      float bias = b2[(wid * 4 + qn) * 16 + nLoc];
#pragma unroll
      for (int mt = 0; mt < MT; ++mt)
        c[qn][mt] = (v8f){bias, bias, bias, bias, bias, bias, bias, bias};
    }
#pragma unroll 4
    for (int kt = 0; kt < 16; ++kt) {
      v16bf a[MT];
#pragma unroll
      for (int mt = 0; mt < MT; ++mt)
        a[mt] = *(const v16bf*)(ldsB[mt] + kt * 512 + lane * 16);
#pragma unroll
      for (int qn = 0; qn < 4; ++qn) {
        const unsigned short* bp =
            pw2 + (kt * 32 + wid * 4 + qn) * 512 + lane * 16;
        v16bf b = *(const v16bf*)bp;
        if (qn == 0 && kt < 15) __builtin_prefetch(bp + 32 * 512, 0, 0);
#pragma unroll
        for (int mt = 0; mt < MT; ++mt)
          c[qn][mt] = wmma_bf16(a[mt], b, c[qn][mt]);
      }
    }
#pragma unroll
    for (int qn = 0; qn < 4; ++qn)
#pragma unroll
      for (int mt = 0; mt < MT; ++mt)
#pragma unroll
        for (int r = 0; r < 8; ++r)
          pack_a_elem(ldsA[mt], r + mBase, (wid * 4 + qn) * 16 + nLoc,
                      fmaxf(c[qn][mt][r], 0.0f));
  }
  __syncthreads();

  // layer 3: N=32 -> 2 N-tiles x MT M-tiles = 8 units, one per wave
  {
    int nt = wid & 1;
    int mt = wid >> 1;            // 0..3
    float bias = b3[nt * 16 + nLoc];
    v8f c = {bias, bias, bias, bias, bias, bias, bias, bias};
#pragma unroll
    for (int kt = 0; kt < 16; ++kt) {
      v16bf a = *(const v16bf*)(ldsA[mt] + kt * 512 + lane * 16);
      v16bf b = *(const v16bf*)(pw3 + (kt * 2 + nt) * 512 + lane * 16);
      c = wmma_bf16(a, b, c);
    }
#pragma unroll
    for (int r = 0; r < 8; ++r) {
      float v = 1.0f / (1.0f + __expf(-c[r]));
      int e = blk * (16 * MT) + mt * 16 + r + mBase;
      y[(size_t)e * NB + nt * 16 + nLoc] = v;
    }
  }
}

// ---------------------------------------------------------------------------
extern "C" void kernel_launch(void* const* d_in, const int* in_sizes, int n_in,
                              void* d_out, int out_size, void* d_ws,
                              size_t ws_size, hipStream_t stream) {
  (void)in_sizes; (void)n_in; (void)out_size; (void)ws_size;
  const float*     objs  = (const float*)d_in[0];
  const long long* ei    = (const long long*)d_in[1];
  const float*     fu_w1 = (const float*)d_in[5];
  const float*     fu_b1 = (const float*)d_in[6];
  const float*     fu_w2 = (const float*)d_in[7];
  const float*     fu_b2 = (const float*)d_in[8];
  const float*     fb_w1 = (const float*)d_in[9];
  const float*     fb_b1 = (const float*)d_in[10];
  const float*     fb_w2 = (const float*)d_in[11];
  const float*     fb_b2 = (const float*)d_in[12];
  const float*     fb_w3 = (const float*)d_in[13];
  const float*     fb_b3 = (const float*)d_in[14];

  float* out = (float*)d_out;
  float* z = out;                                   // N x 16
  float* x = z + (size_t)N_NODES * NU;              // N x 256 (copy of objs)
  float* y = x + (size_t)N_NODES * W_FEAT;          // E x 32

  unsigned short* ws     = (unsigned short*)d_ws;
  unsigned short* objsB  = ws;                            // 12.8M bf16
  unsigned short* pw_fu1 = objsB + (size_t)N_NODES * W_FEAT;
  unsigned short* pw_fu2 = pw_fu1 + 256 * 512;
  unsigned short* pw_fb1 = pw_fu2 + 512 * 16;
  unsigned short* pw_fb2 = pw_fb1 + 512 * 512;
  unsigned short* pw_fb3 = pw_fb2 + 512 * 512;

  const int nObjsElems = N_NODES * W_FEAT;                // 12,800,000
  k_cvt_objs<<<nObjsElems / 1024, 256, 0, stream>>>(objs, objsB, nObjsElems);
  k_pack_w<<<(256 * 512 + 255) / 256, 256, 0, stream>>>(fu_w1, pw_fu1, 256, 512);
  k_pack_w<<<(512 * 16 + 255) / 256, 256, 0, stream>>>(fu_w2, pw_fu2, 512, 16);
  k_pack_w<<<(512 * 512 + 255) / 256, 256, 0, stream>>>(fb_w1, pw_fb1, 512, 512);
  k_pack_w<<<(512 * 512 + 255) / 256, 256, 0, stream>>>(fb_w2, pw_fb2, 512, 512);
  k_pack_w<<<(512 * 32 + 255) / 256, 256, 0, stream>>>(fb_w3, pw_fb3, 512, 32);

  hipMemcpyAsync(x, objs, (size_t)N_NODES * W_FEAT * sizeof(float),
                 hipMemcpyDeviceToDevice, stream);

  k_node_mlp<<<N_NODES / 16, 128, 0, stream>>>(objsB, pw_fu1, fu_b1, pw_fu2,
                                               fu_b2, z);
  k_edge_mlp<<<E_EDGES / (16 * MT), 256, 0, stream>>>(objsB, ei, pw_fb1, fb_b1,
                                                      pw_fb2, fb_b2, pw_fb3,
                                                      fb_b3, y);
}
